// DifferentialWordSegmentation_31147102831164
// MI455X (gfx1250) — matrix operand order repaired and checked by reading, not compiled, not run on hardware
//
#include <hip/hip_runtime.h>
#include <hip/hip_bf16.h>
#include <math.h>

// Problem constants (match reference)
#define BB   16
#define NN   2048
#define HH   256
#define LL   2047      // N-1
#define MW   256       // MAX_WORDS
#define NP   2064      // N padded by 16 zero rows (for shifted GEMM tail)

typedef __attribute__((ext_vector_type(16))) __bf16        v16bf;
typedef __attribute__((ext_vector_type(8)))  float         v8f;
typedef __attribute__((ext_vector_type(8)))  unsigned int  v8u;
typedef __attribute__((ext_vector_type(4)))  unsigned int  v4u;
typedef __attribute__((ext_vector_type(8)))  int           v8i;
typedef __attribute__((ext_vector_type(4)))  int           v4i;

#if defined(__has_builtin)
#  if __has_builtin(__builtin_amdgcn_tensor_load_to_lds) && \
      __has_builtin(__builtin_amdgcn_s_wait_tensorcnt)
#    define USE_TDM 1
#  endif
#endif
#ifndef USE_TDM
#  define USE_TDM 0
#endif

__device__ __forceinline__ float reluf(float x) { return fmaxf(x, 0.0f); }

__device__ __forceinline__ float gelu_tanh(float x) {
    float x3 = x * x * x;
    return 0.5f * x * (1.0f + tanhf(0.7978845608028654f * (x + 0.044715f * x3)));
}

// fp32 -> bf16 bits, round-to-nearest-even
__device__ __forceinline__ unsigned short f2bf(float f) {
    unsigned int x = __float_as_uint(f);
    unsigned int r = x + 0x7FFFu + ((x >> 16) & 1u);
    return (unsigned short)(r >> 16);
}

// ---- WMMA fragment loaders (contiguous-K order k = 16*half + s, wave32) ----
// A 16x32 bf16: lane[0..15] row M=lane, K {0..15}; lane[16..31] row M=lane-16, K {16..31}.
__device__ __forceinline__ v16bf load_a_frag(const unsigned short* A, int ld,
                                             int row0, int k0, int lane) {
    int half = lane >> 4, m = lane & 15;
    const unsigned short* rp = A + (size_t)(row0 + m) * ld + k0 + half * 16;
    v8u r;
#pragma unroll
    for (int i = 0; i < 8; ++i) {
        int kb = 2 * i;
        r[i] = *(const unsigned int*)(rp + kb);
    }
    return __builtin_bit_cast(v16bf, r);
}

// B 32x16 bf16 staged column-major (Bcm[col*ldk + k]): lane[0..15] col N=lane, VGPR r holds
// K=2r,2r+1; lane[16..31] col N=lane-16, K=16+2r,16+2r+1.
__device__ __forceinline__ v16bf load_b_frag(const unsigned short* Bcm, int ldk,
                                             int k0, int col0, int lane) {
    int half = lane >> 4, c = lane & 15;
    const unsigned short* cp = Bcm + (size_t)(col0 + c) * ldk + k0 + half * 16;
    v8u r;
#pragma unroll
    for (int i = 0; i < 8; ++i) r[i] = *(const unsigned int*)(cp + 2 * i);
    return __builtin_bit_cast(v16bf, r);
}

__device__ __forceinline__ v8f wmma_bf16(v16bf a, v16bf b, v8f c) {
    return __builtin_amdgcn_wmma_f32_16x16x32_bf16(false, a, false, b, (short)0, c,
                                                   false, false);
}

#if USE_TDM
// Issue a TDM 2-D tile load: tensor is the col-major bf16 weight buffer
// (contiguous dim0 = K of length 256, stride 256; dim1 = 512 columns).
// Tile = 32 (K) x 512 (cols) starting at gptr (= base + k0 elements), landing
// contiguously in LDS as [col][k_local] with ldk = 32.  ISA ch.8 D# layout.
__device__ __forceinline__ void tdm_load_slab(unsigned lds_off, const unsigned short* gptr) {
    unsigned long long ga = (unsigned long long)(size_t)gptr;
    v4u g0;
    g0[0] = 1u;                                        // count=1 valid descriptor
    g0[1] = lds_off;                                   // lds_addr
    g0[2] = (unsigned)ga;                              // global_addr[31:0]
    g0[3] = (unsigned)((ga >> 32) & 0x1FFFFFFull) | (2u << 30);  // addr[56:32] | type=2
    v8i g1;
    g1[0] = (int)(1u << 16);                           // data_size = 1 -> 2 bytes
    g1[1] = (int)((256u & 0xFFFFu) << 16);             // tensor_dim0 = 256 (lo)
    g1[2] = (int)(((256u >> 16) & 0xFFFFu) | ((512u & 0xFFFFu) << 16)); // dim0 hi | dim1 lo
    g1[3] = (int)(((512u >> 16) & 0xFFFFu) | (32u << 16));              // dim1 hi | tile_dim0=32
    g1[4] = (int)512;                                  // tile_dim1 = 512, tile_dim2 = 0
    g1[5] = (int)256;                                  // tensor_dim0_stride = 256 (lo)
    g1[6] = 0;                                         // stride hi | dim1_stride lo (2-D: unused)
    g1[7] = 0;
    v4i z4 = {0, 0, 0, 0};
#if __clang_major__ >= 23
    v8i z8 = {0, 0, 0, 0, 0, 0, 0, 0};
    __builtin_amdgcn_tensor_load_to_lds(g0, g1, z4, z4, z8, 0);
#else
    __builtin_amdgcn_tensor_load_to_lds(g0, g1, z4, z4, 0);
#endif
}
#endif

// ---------------- Kernel A: l2norm rows -> bf16 (padded to NP rows/batch) ----------------
__global__ void k_normalize(const float* __restrict__ x, unsigned short* __restrict__ A) {
    int wave = threadIdx.x >> 5, lane = threadIdx.x & 31;
    const int per = NP / 8;
    int b = blockIdx.x / per;
    int n = (blockIdx.x % per) * 8 + wave;
    unsigned short* out = A + ((size_t)b * NP + n) * HH;
    if (n >= NN) {                                // zero pad rows (wave-uniform branch)
#pragma unroll
        for (int j = 0; j < 8; ++j) out[lane + 32 * j] = 0;
        return;
    }
    const float* row = x + ((size_t)b * NN + n) * HH;
    float v[8]; float ss = 0.f;
#pragma unroll
    for (int j = 0; j < 8; ++j) { v[j] = row[lane + 32 * j]; ss += v[j] * v[j]; }
#pragma unroll
    for (int off = 16; off > 0; off >>= 1) ss += __shfl_xor(ss, off, 32);
    float rn = rsqrtf(ss);
#pragma unroll
    for (int j = 0; j < 8; ++j) out[lane + 32 * j] = f2bf(v[j] * rn);
}

// ---------------- Kernel W: pack W1 (top|bot -> 512 cols) and Wt as col-major bf16 ----------------
__global__ void k_convw(const float* __restrict__ W1, const float* __restrict__ Wt,
                        unsigned short* __restrict__ W12CM,
                        unsigned short* __restrict__ WtCM) {
    int i = blockIdx.x * blockDim.x + threadIdx.x;   // 0..65535
    int k = i >> 8, c = i & 255;
    W12CM[(size_t)c * 256 + k]         = f2bf(W1[(size_t)k * 256 + c]);          // top: cols 0..255
    W12CM[(size_t)(c + 256) * 256 + k] = f2bf(W1[(size_t)(k + 256) * 256 + c]);  // bot: cols 256..511
    WtCM [(size_t)c * 256 + k]         = f2bf(Wt[(size_t)k * 256 + c]);
}

// ---------------- Kernel B: fused shifted GEMM + gelu + dot(W2) + sigmoid -> S ----------------
// C[n,:] = Anorm[n,:]@W1top + Anorm[n+1,:]@W1bot ; S[n] = sigmoid( gelu(C[n]+b1) . W2 )
// Weight panels streamed into LDS by the Tensor Data Mover, double-buffered, and
// consumed by v_wmma_f32_16x16x32_bf16.
__global__ void k_gemm_sim(const unsigned short* __restrict__ A,
                           const unsigned short* __restrict__ W12CM,
                           const float* __restrict__ b1, const float* __restrict__ W2,
                           float* __restrict__ S) {
    int b = blockIdx.y, n0 = blockIdx.x * 32;
    int wave = threadIdx.x >> 5, lane = threadIdx.x & 31;
    int rt = wave & 1, cg = wave >> 1;              // row tile, col group (4 tiles each)
    const unsigned short* Ab = A + (size_t)b * NP * HH;
    v8f acc[4] = {};
    int rowA = n0 + 16 * rt;
#if USE_TDM
    __shared__ unsigned short sh[2][32 * 512];      // 2 x 32KB K-slabs (cols 0..255 top, 256..511 bot)
    if (wave == 0) tdm_load_slab((unsigned)(size_t)&sh[0][0], W12CM);
    for (int s = 0; s < 8; ++s) {
        int cur = s & 1;
        int k0 = s * 32;
        if (wave == 0) __builtin_amdgcn_s_wait_tensorcnt(0);   // slab s landed
        __syncthreads();
        if (wave == 0 && s < 7)                                 // prefetch slab s+1 via TDM
            tdm_load_slab((unsigned)(size_t)&sh[cur ^ 1][0], W12CM + (k0 + 32));
        v16bf a0 = load_a_frag(Ab, HH, rowA,     k0, lane);
        v16bf a1 = load_a_frag(Ab, HH, rowA + 1, k0, lane);
        const unsigned short* slab = &sh[cur][0];               // [col][k_local], ldk=32
#pragma unroll
        for (int i = 0; i < 4; ++i) {
            int col0 = (cg * 4 + i) * 16;
            v16bf bt = load_b_frag(slab, 32, 0, col0,       lane);
            acc[i] = wmma_bf16(a0, bt, acc[i]);
            v16bf bb = load_b_frag(slab, 32, 0, col0 + 256, lane);
            acc[i] = wmma_bf16(a1, bb, acc[i]);
        }
    }
#else
    __builtin_prefetch(W12CM + (size_t)(cg * 64) * 256, 0, 0);
    for (int k0 = 0; k0 < 256; k0 += 32) {
        v16bf a0 = load_a_frag(Ab, HH, rowA,     k0, lane);
        v16bf a1 = load_a_frag(Ab, HH, rowA + 1, k0, lane);
#pragma unroll
        for (int i = 0; i < 4; ++i) {
            int col0 = (cg * 4 + i) * 16;
            v16bf bt = load_b_frag(W12CM, 256, k0, col0,       lane);
            acc[i] = wmma_bf16(a0, bt, acc[i]);
            v16bf bb = load_b_frag(W12CM, 256, k0, col0 + 256, lane);
            acc[i] = wmma_bf16(a1, bb, acc[i]);
        }
    }
#endif
    __shared__ float Ssum[32];
    if (threadIdx.x < 32) Ssum[threadIdx.x] = 0.f;
    __syncthreads();
    int half = lane >> 4, nl = lane & 15;
    float part[8];
#pragma unroll
    for (int r = 0; r < 8; ++r) part[r] = 0.f;
#pragma unroll
    for (int i = 0; i < 4; ++i) {
        int col = (cg * 4 + i) * 16 + nl;
        float bias = b1[col], w2 = W2[col];
#pragma unroll
        for (int r = 0; r < 8; ++r) part[r] += gelu_tanh(acc[i][r] + bias) * w2;
    }
#pragma unroll
    for (int r = 0; r < 8; ++r) atomicAdd(&Ssum[16 * rt + r + 8 * half], part[r]);
    __syncthreads();
    if (threadIdx.x < 32) {
        int n = n0 + threadIdx.x;
        if (n < LL) S[(size_t)b * NN + n] = 1.0f / (1.0f + expf(-Ssum[threadIdx.x]));
    }
}

// ---------------- Kernel C: per-batch min/max of S ----------------
__global__ void k_minmax(const float* __restrict__ S, float* __restrict__ mm) {
    int b = blockIdx.x, t = threadIdx.x;
    __shared__ float smn[256], smx[256];
    float mn = 3.4e38f, mx = -3.4e38f;
    for (int i = t; i < LL; i += 256) {
        float v = S[(size_t)b * NN + i];
        mn = fminf(mn, v); mx = fmaxf(mx, v);
    }
    smn[t] = mn; smx[t] = mx; __syncthreads();
    for (int o = 128; o > 0; o >>= 1) {
        if (t < o) { smn[t] = fminf(smn[t], smn[t + o]); smx[t] = fmaxf(smx[t], smx[t + o]); }
        __syncthreads();
    }
    if (t == 0) { mm[b * 2] = smn[0]; mm[b * 2 + 1] = smx[0]; }
}

// ---------------- Kernel D: D-normalization + peak detector -> P (B,N) ----------------
__global__ void k_peaks(const float* __restrict__ S, const float* __restrict__ mm,
                        const float* __restrict__ phn, float* __restrict__ P) {
    int b = blockIdx.y;
    int n = blockIdx.x * 256 + threadIdx.x;
    float smin = mm[b * 2];
    float inv  = 1.0f / (mm[b * 2 + 1] - smin);
    const float* Sb = S + (size_t)b * NN;
    auto Dv = [&](int i) { return 1.0f - (Sb[i] - smin) * inv; };
    float p = 0.f;
    if (n < LL) {
        float d = Dv(n);
        float fo, so;
        if (n == 0)            fo = reluf(d - Dv(1));
        else if (n <= LL - 3)  fo = fminf(reluf(d - Dv(n - 1)), reluf(d - Dv(n + 1)));
        else                   fo = reluf(d - Dv(n - 2));   // the fo[:,-2:] overwrite
        if (n <= 1)            so = reluf(d - Dv(n + 2));
        else if (n <= LL - 3)  so = fminf(reluf(d - Dv(n - 2)), reluf(d - Dv(n + 2)));
        else                   so = 0.f;
        p = fminf(reluf(fmaxf(fo, so) - 0.01f), fo);
    }
    P[(size_t)b * NN + n] = reluf(p + phn[(size_t)b * NN + n]);
}

// ---------------- Kernel E: Gumbel scores, bitonic top-k, segment starts/counts ----------------
__global__ void k_select(const float* __restrict__ P, const int* __restrict__ wl,
                         int* __restrict__ starts, int* __restrict__ counts) {
    __shared__ float key[NN];
    __shared__ int   val[NN];
    __shared__ int   flg[NN];
    __shared__ int   cs[NN];
    __shared__ int   tot[256];
    __shared__ int   pre[257];
    __shared__ int   st[257];
    int b = blockIdx.x, t = threadIdx.x;
    for (int n = t; n < NN; n += 256) {
        unsigned int h = (unsigned)n * 0x9E3779B9u ^ ((unsigned)b * 0x85EBCA6Bu + 0xC2B2AE35u);
        h ^= h >> 16; h *= 0x7FEB352Du; h ^= h >> 15; h *= 0x846CA68Bu; h ^= h >> 16;
        float u = ((float)h + 0.5f) * (1.0f / 4294967296.0f);
        float g = -logf(-logf(u));                       // Gumbel(0,1), deterministic hash PRNG
        key[n] = logf(P[(size_t)b * NN + n] + 1e-6f) + g;
        val[n] = n;
    }
    __syncthreads();
    // bitonic sort, descending by key, ascending index on ties (stable argsort(-scores))
    for (int k = 2; k <= NN; k <<= 1) {
        for (int j = k >> 1; j > 0; j >>= 1) {
            for (int i = t; i < NN; i += 256) {
                int l = i ^ j;
                if (l > i) {
                    bool des = ((i & k) == 0);
                    float ki = key[i], kl = key[l];
                    int vi = val[i], vl = val[l];
                    bool sw = des ? (ki < kl || (ki == kl && vi > vl))
                                  : (ki > kl || (ki == kl && vi < vl));
                    if (sw) { key[i] = kl; key[l] = ki; val[i] = vl; val[l] = vi; }
                }
            }
            __syncthreads();
        }
    }
    int L = wl[b];
    for (int n = t; n < NN; n += 256) flg[n] = 0;
    __syncthreads();
    for (int r = t; r < L; r += 256) flg[val[r]] = 1;    // Pb = top-L positions
    __syncthreads();
    // inclusive cumsum of flg (segment ids)
    int base = t * 8, s = 0;
#pragma unroll
    for (int j = 0; j < 8; ++j) { s += flg[base + j]; cs[base + j] = s; }
    tot[t] = s; __syncthreads();
    if (t == 0) { int a = 0; pre[0] = 0; for (int i = 0; i < 256; ++i) { a += tot[i]; pre[i + 1] = a; } }
    __syncthreads();
    int off = pre[t];
#pragma unroll
    for (int j = 0; j < 8; ++j) cs[base + j] += off;
    __syncthreads();
    for (int n = t; n < NN; n += 256) if (flg[n]) st[cs[n] - 1] = n;
    if (t == 0) st[L] = NN;
    __syncthreads();
    for (int m = t; m < MW; m += 256) {
        int c = (m < L) ? (st[m + 1] - st[m]) : 0;
        counts[b * MW + m] = c;
        starts[b * MW + m] = (m < L) ? st[m] : 0;
    }
}

// ---------------- Kernel F: segment mean pooling -> bf16 pooled (B,M,H) ----------------
__global__ void k_pool(const float* __restrict__ x, const int* __restrict__ starts,
                       const int* __restrict__ counts, unsigned short* __restrict__ pooled) {
    int b = blockIdx.y, m = blockIdx.x, h = threadIdx.x;
    int c = counts[b * MW + m], s = starts[b * MW + m];
    const float* xb = x + (size_t)b * NN * HH;
    float acc = 0.f;
    for (int q = 0; q < c; ++q) acc += xb[(size_t)(s + q) * HH + h];
    float v = (c > 0) ? acc / (float)c : 0.f;
    pooled[((size_t)b * MW + m) * HH + h] = f2bf(v);
}

// ---------------- Kernel G: out[b] = pooled[b] @ Wt  (bf16 WMMA, f32 out) ----------------
__global__ void k_gemm_out(const unsigned short* __restrict__ pooled,
                           const unsigned short* __restrict__ WtCM,
                           float* __restrict__ out) {
    int b = blockIdx.y, n0 = blockIdx.x * 32;
    int wave = threadIdx.x >> 5, lane = threadIdx.x & 31;
    int rt = wave & 1, cg = wave >> 1;
    const unsigned short* Ab = pooled + (size_t)b * MW * HH;
    __builtin_prefetch(WtCM + (size_t)(cg * 64) * 256, 0, 0);
    v8f acc[4] = {};
    int rowA = n0 + 16 * rt;
    for (int k0 = 0; k0 < 256; k0 += 32) {
        v16bf a = load_a_frag(Ab, HH, rowA, k0, lane);
#pragma unroll
        for (int i = 0; i < 4; ++i) {
            int col0 = (cg * 4 + i) * 16;
            v16bf bf = load_b_frag(WtCM, 256, k0, col0, lane);
            acc[i] = wmma_bf16(a, bf, acc[i]);
        }
    }
    int half = lane >> 4, nl = lane & 15;
    float* ob = out + (size_t)b * MW * HH;
#pragma unroll
    for (int i = 0; i < 4; ++i) {
        int col0 = (cg * 4 + i) * 16;
#pragma unroll
        for (int r = 0; r < 8; ++r) {
            int row = rowA + r + 8 * half;
            ob[(size_t)row * HH + col0 + nl] = acc[i][r];
        }
    }
}

extern "C" void kernel_launch(void* const* d_in, const int* in_sizes, int n_in,
                              void* d_out, int out_size, void* d_ws, size_t ws_size,
                              hipStream_t stream) {
    (void)in_sizes; (void)n_in; (void)out_size; (void)ws_size;
    const float* seg = (const float*)d_in[0];   // (B,N,H)
    const float* phn = (const float*)d_in[1];   // (B,N)
    const int*   wl  = (const int*)  d_in[2];   // (B,)
    const float* W1  = (const float*)d_in[3];   // (2H,H)
    const float* b1  = (const float*)d_in[4];   // (H,)
    const float* W2  = (const float*)d_in[5];   // (H,)
    const float* Wt  = (const float*)d_in[6];   // (H,H)
    float* out = (float*)d_out;                 // (B,M,H)

    // carve workspace (~20 MB total)
    char* base = (char*)d_ws;
    size_t off = 0;
    auto carve = [&](size_t bytes) { off = (off + 255) & ~(size_t)255; char* p = base + off; off += bytes; return p; };
    unsigned short* A      = (unsigned short*)carve((size_t)BB * NP * HH * 2);
    unsigned short* W12CM  = (unsigned short*)carve((size_t)512 * 256 * 2);   // top|bot packed
    unsigned short* WtCM   = (unsigned short*)carve((size_t)HH * HH * 2);
    float*          S      = (float*)carve((size_t)BB * NN * 4);
    float*          mm     = (float*)carve((size_t)BB * 2 * 4);
    float*          Pbuf   = (float*)carve((size_t)BB * NN * 4);
    int*            starts = (int*)carve((size_t)BB * MW * 4);
    int*            counts = (int*)carve((size_t)BB * MW * 4);
    unsigned short* pooled = (unsigned short*)carve((size_t)BB * MW * HH * 2);

    k_normalize<<<dim3(BB * (NP / 8)), dim3(256), 0, stream>>>(seg, A);
    k_convw<<<dim3(256), dim3(256), 0, stream>>>(W1, Wt, W12CM, WtCM);
    k_gemm_sim<<<dim3(NN / 32, BB), dim3(256), 0, stream>>>(A, W12CM, b1, W2, S);
    k_minmax<<<dim3(BB), dim3(256), 0, stream>>>(S, mm);
    k_peaks<<<dim3(NN / 256, BB), dim3(256), 0, stream>>>(S, mm, phn, Pbuf);
    k_select<<<dim3(BB), dim3(256), 0, stream>>>(Pbuf, wl, starts, counts);
    k_pool<<<dim3(MW, BB), dim3(256), 0, stream>>>(seg, starts, counts, pooled);
    k_gemm_out<<<dim3(MW / 32, BB), dim3(256), 0, stream>>>(pooled, WtCM, out);
}
